// Model_60567628808197
// MI455X (gfx1250) — compile-verified, matrix operand order checked by reading, not run on hardware
//
#include <hip/hip_runtime.h>
#include <hip/hip_bf16.h>
#include <cmath>

// ---------------------------------------------------------------------------
// Model constants (from the reference)
// ---------------------------------------------------------------------------
#define BATCH      4
#define L_ENC      2048
#define L_DEC      1024
#define PRED_LEN   512
#define D_MODEL    512
#define N_HEADS    8
#define D_HEAD     64
#define D_FF       2048
#define FACTOR     5
#define C_IN       7
#define C_OUT      7

typedef __attribute__((ext_vector_type(16))) _Float16 v16h;
typedef __attribute__((ext_vector_type(8)))  float    v8f;

union U16h { v16h v; _Float16 h[16]; };
union U8f  { v8f  v; float    f[8];  };

// ---------------------------------------------------------------------------
// WMMA GEMM:  C[M,N] = act( A[M,K] @ W[N,K]^T + bias[N] )
//   ACT: 0 = none, 1 = tanh-GELU (compile-time -> branch-free epilogue)
// Block: 256 threads (8 wave32), tile BM=128 x BN=128, K-step 32.
// Each wave computes a 32x64 patch as 2x4 v_wmma_f32_16x16x32_f16 tiles.
// Double-buffered LDS + software pipeline: next K-tile's 8 global_load_b128
// are issued before the current tile's ds_loads/WMMAs, so global latency
// hides behind matrix math; conversion+ds_store happens after the WMMAs.
// ---------------------------------------------------------------------------
#define BM 128
#define BN 128
#define BKK 32

template <int ACT>
__global__ __launch_bounds__(256) void wmma_gemm_kernel(
    const float* __restrict__ A, const float* __restrict__ W,
    const float* __restrict__ bias, float* __restrict__ C,
    int M, int N, int K)
{
  __shared__ _Float16 As[2][BM][BKK + 8];
  __shared__ _Float16 Bs[2][BN][BKK + 8];

  const int tid  = threadIdx.x;
  const int lane = tid & 31;
  const int wave = tid >> 5;     // 0..7
  const int wm   = wave & 3;     // 4 waves along M (32 rows each)
  const int wn   = wave >> 2;    // 2 waves along N (64 cols each)

  const int m0 = blockIdx.y * BM;
  const int n0 = blockIdx.x * BN;

  v8f acc[2][4];
#pragma unroll
  for (int mi = 0; mi < 2; ++mi)
#pragma unroll
    for (int ni = 0; ni < 4; ++ni) acc[mi][ni] = (v8f){};

  const int lr    = lane & 15;
  const int khalf = (lane < 16) ? 0 : 8;   // §7.12.2 16-bit A layout half-split

  const int ldr = tid >> 3;          // 0..31
  const int ldc = (tid & 7) * 4;     // 0,4,...,28

  // row base pointers, bumped by BKK floats per K-step
  const float* ap[4];
  const float* bp[4];
#pragma unroll
  for (int rr = 0; rr < 4; ++rr) {
    ap[rr] = A + (size_t)(m0 + ldr + rr * 32) * K + ldc;
    bp[rr] = W + (size_t)(n0 + ldr + rr * 32) * K + ldc;
  }

  // ---- preload tile 0: 8 loads in flight, then convert+store ----
  float4 va[4], vb[4];
#pragma unroll
  for (int rr = 0; rr < 4; ++rr) {
    va[rr] = *reinterpret_cast<const float4*>(ap[rr]);
    vb[rr] = *reinterpret_cast<const float4*>(bp[rr]);
    ap[rr] += BKK; bp[rr] += BKK;
  }
#pragma unroll
  for (int rr = 0; rr < 4; ++rr) {
    const int r = ldr + rr * 32;
    As[0][r][ldc + 0] = (_Float16)va[rr].x;
    As[0][r][ldc + 1] = (_Float16)va[rr].y;
    As[0][r][ldc + 2] = (_Float16)va[rr].z;
    As[0][r][ldc + 3] = (_Float16)va[rr].w;
    Bs[0][r][ldc + 0] = (_Float16)vb[rr].x;
    Bs[0][r][ldc + 1] = (_Float16)vb[rr].y;
    Bs[0][r][ldc + 2] = (_Float16)vb[rr].z;
    Bs[0][r][ldc + 3] = (_Float16)vb[rr].w;
  }
  __syncthreads();

  const int nk = K / BKK;
  for (int kt = 0; kt < nk; ++kt) {
    const int cur = kt & 1, nxt = cur ^ 1;
    const bool more = (kt + 1 < nk);

    // ---- issue next tile's global loads (latency hidden by WMMAs below) ----
    if (more) {
#pragma unroll
      for (int rr = 0; rr < 4; ++rr) {
        va[rr] = *reinterpret_cast<const float4*>(ap[rr]);
        vb[rr] = *reinterpret_cast<const float4*>(bp[rr]);
        ap[rr] += BKK; bp[rr] += BKK;
      }
    }

    // ---- build fragments per §7.12.2 (16-bit A 16x32 layout) ----
    U16h af[2], bf[4];
#pragma unroll
    for (int mi = 0; mi < 2; ++mi) {
      const int row = wm * 32 + mi * 16 + lr;
#pragma unroll
      for (int e = 0; e < 8; ++e) {
        af[mi].h[e]     = As[cur][row][khalf + e];
        af[mi].h[8 + e] = As[cur][row][16 + khalf + e];
      }
    }
#pragma unroll
    for (int ni = 0; ni < 4; ++ni) {
      const int col = wn * 64 + ni * 16 + lr;
#pragma unroll
      for (int e = 0; e < 8; ++e) {
        bf[ni].h[e]     = Bs[cur][col][khalf + e];
        bf[ni].h[8 + e] = Bs[cur][col][16 + khalf + e];
      }
    }

#pragma unroll
    for (int mi = 0; mi < 2; ++mi)
#pragma unroll
      for (int ni = 0; ni < 4; ++ni)
        acc[mi][ni] = __builtin_amdgcn_wmma_f32_16x16x32_f16(
            false, af[mi].v, false, bf[ni].v, (short)0, acc[mi][ni],
            false, false);

    // ---- convert + store next tile into the other LDS buffer ----
    if (more) {
#pragma unroll
      for (int rr = 0; rr < 4; ++rr) {
        const int r = ldr + rr * 32;
        As[nxt][r][ldc + 0] = (_Float16)va[rr].x;
        As[nxt][r][ldc + 1] = (_Float16)va[rr].y;
        As[nxt][r][ldc + 2] = (_Float16)va[rr].z;
        As[nxt][r][ldc + 3] = (_Float16)va[rr].w;
        Bs[nxt][r][ldc + 0] = (_Float16)vb[rr].x;
        Bs[nxt][r][ldc + 1] = (_Float16)vb[rr].y;
        Bs[nxt][r][ldc + 2] = (_Float16)vb[rr].z;
        Bs[nxt][r][ldc + 3] = (_Float16)vb[rr].w;
      }
    }
    __syncthreads();
  }

  // ---- epilogue: bias + optional GELU, branch-free ----
#pragma unroll
  for (int ni = 0; ni < 4; ++ni) {
    const int nIdx = n0 + wn * 64 + ni * 16 + lr;
    const float bv = bias[nIdx];
#pragma unroll
    for (int mi = 0; mi < 2; ++mi) {
      U8f cu; cu.v = acc[mi][ni];
#pragma unroll
      for (int r = 0; r < 8; ++r) {
        const int mIdx = m0 + wm * 32 + mi * 16 + ((lane < 16) ? r : (8 + r));
        float v = cu.f[r] + bv;
        if (ACT == 1) {
          const float x3 = v * v * v;
          v = 0.5f * v * (1.f + tanhf(0.7978845608028654f * (v + 0.044715f * x3)));
        }
        C[(size_t)mIdx * N + nIdx] = v;
      }
    }
  }
}

// ---------------------------------------------------------------------------
// Embedding: circular 3-tap conv over channels + state embedding + sinusoid PE
// ---------------------------------------------------------------------------
__global__ void embed_kernel(const int* __restrict__ xstate,
                             const float* __restrict__ x,
                             const float* __restrict__ convw,
                             const float* __restrict__ stemb,
                             float* __restrict__ out, int L)
{
  const int idx = blockIdx.x * blockDim.x + threadIdx.x;
  const int total = BATCH * L * D_MODEL;
  if (idx >= total) return;
  const int d = idx & (D_MODEL - 1);
  const int l = (idx / D_MODEL) % L;
  const int b = idx / (D_MODEL * L);

  float v = 0.f;
#pragma unroll
  for (int k = 0; k < 3; ++k) {
    const int ls = (l + k - 1 + L) % L;   // wrap padding
#pragma unroll
    for (int c = 0; c < C_IN; ++c)
      v += x[((size_t)b * L + ls) * C_IN + c] * convw[d * (C_IN * 3) + c * 3 + k];
  }
  v += stemb[xstate[(size_t)b * L + l] * D_MODEL + d];

  const int pair = d >> 1;
  const float freq = __expf(-(float)(2 * pair) * (9.210340371976184f / 512.f));
  const float ang = (float)l * freq;
  v += (d & 1) ? cosf(ang) : sinf(ang);

  out[(size_t)(b * L + l) * D_MODEL + d] = v;
}

// ---------------------------------------------------------------------------
// LayerNorm over D_MODEL with optional fused residual: out = LN(X [+ R])*g + b
// One wave32 per row, 16 elements per lane.
// ---------------------------------------------------------------------------
__global__ __launch_bounds__(256) void layernorm_kernel(
    const float* __restrict__ X, const float* __restrict__ R,
    const float* __restrict__ g, const float* __restrict__ bta,
    float* __restrict__ out, int rows)
{
  const int row  = blockIdx.x * 8 + (threadIdx.x >> 5);
  const int lane = threadIdx.x & 31;
  if (row >= rows) return;
  const float* xr = X + (size_t)row * D_MODEL;
  const float* rr = R ? (R + (size_t)row * D_MODEL) : nullptr;
  float vals[16];
  float s = 0.f, ss = 0.f;
#pragma unroll
  for (int i = 0; i < 16; ++i) {
    float v = xr[lane + i * 32];
    if (rr) v += rr[lane + i * 32];
    vals[i] = v; s += v; ss += v * v;
  }
#pragma unroll
  for (int off = 16; off > 0; off >>= 1) {
    s  += __shfl_xor(s,  off, 32);
    ss += __shfl_xor(ss, off, 32);
  }
  const float mean = s * (1.f / D_MODEL);
  const float var  = ss * (1.f / D_MODEL) - mean * mean;
  const float inv  = rsqrtf(var + 1e-5f);
#pragma unroll
  for (int i = 0; i < 16; ++i) {
    const int c = lane + i * 32;
    out[(size_t)row * D_MODEL + c] = (vals[i] - mean) * inv * g[c] + bta[c];
  }
}

// ---------------------------------------------------------------------------
// ProbSparse attention helpers
// Activations live in (B, L, H*D) layout; head h occupies columns h*64..h*64+63.
// ---------------------------------------------------------------------------
__device__ __forceinline__ unsigned hash_u32(unsigned x) {
  x ^= x >> 16; x *= 0x7feb352du; x ^= x >> 15; x *= 0x846ca68bu; x ^= x >> 16;
  return x;
}

// context init (non-masked): broadcast mean of V over keys
__global__ __launch_bounds__(256) void ctx_mean_kernel(
    const float* __restrict__ Vb, float* __restrict__ CTX, int Lq, int Lk)
{
  const int bh = blockIdx.x, b = bh / N_HEADS, h = bh % N_HEADS;
  const int t = threadIdx.x;
  __shared__ float md[D_HEAD];
  if (t < D_HEAD) {
    float s = 0.f;
    for (int k = 0; k < Lk; ++k)
      s += Vb[((size_t)b * Lk + k) * D_MODEL + h * D_HEAD + t];
    md[t] = s / (float)Lk;
  }
  __syncthreads();
  for (int idx = t; idx < Lq * D_HEAD; idx += 256) {
    const int l = idx >> 6, d = idx & 63;
    CTX[((size_t)b * Lq + l) * D_MODEL + h * D_HEAD + d] = md[d];
  }
}

// context init (masked/self): causal cumsum of V
__global__ __launch_bounds__(64) void ctx_cumsum_kernel(
    const float* __restrict__ Vb, float* __restrict__ CTX, int L)
{
  const int bh = blockIdx.x, b = bh / N_HEADS, h = bh % N_HEADS;
  const int d = threadIdx.x;
  float a = 0.f;
  for (int l = 0; l < L; ++l) {
    a += Vb[((size_t)b * L + l) * D_MODEL + h * D_HEAD + d];
    CTX[((size_t)b * L + l) * D_MODEL + h * D_HEAD + d] = a;
  }
}

// sparsity measure M = max - mean over sampled keys, then iterative top-u
__global__ __launch_bounds__(256) void prob_topk_kernel(
    const float* __restrict__ Qb, const float* __restrict__ Kb,
    int* __restrict__ Mtop, int Lq, int Lk, int Upart, int u, unsigned salt)
{
  const int bh = blockIdx.x, b = bh / N_HEADS, h = bh % N_HEADS;
  const int tid = threadIdx.x;
  __shared__ float Mv[2048];
  __shared__ float rv[256];
  __shared__ int   ri[256];

  for (int l = tid; l < Lq; l += 256) {
    float q[D_HEAD];
#pragma unroll
    for (int d = 0; d < D_HEAD; ++d)
      q[d] = Qb[((size_t)b * Lq + l) * D_MODEL + h * D_HEAD + d];
    float mx = -3.4e38f, sm = 0.f;
    for (int j = 0; j < Upart; ++j) {
      const unsigned kidx =
          hash_u32(salt * 0x9E3779B9u + (unsigned)(l * Upart + j)) % (unsigned)Lk;
      float dot = 0.f;
#pragma unroll
      for (int d = 0; d < D_HEAD; ++d)
        dot += q[d] * Kb[((size_t)b * Lk + kidx) * D_MODEL + h * D_HEAD + d];
      mx = fmaxf(mx, dot); sm += dot;
    }
    Mv[l] = mx - sm / (float)Lk;
  }
  __syncthreads();

  for (int i = 0; i < u; ++i) {
    float bv = -3.4e38f; int bi = Lq;
    for (int l = tid; l < Lq; l += 256) {
      const float v = Mv[l];
      if (v > bv || (v == bv && l < bi)) { bv = v; bi = l; }
    }
    rv[tid] = bv; ri[tid] = bi;
    for (int off = 128; off > 0; off >>= 1) {
      __syncthreads();
      if (tid < off) {
        const float ov = rv[tid + off]; const int oi = ri[tid + off];
        if (ov > rv[tid] || (ov == rv[tid] && oi < ri[tid])) { rv[tid] = ov; ri[tid] = oi; }
      }
    }
    __syncthreads();
    if (tid == 0) { Mtop[bh * u + i] = ri[0]; Mv[ri[0]] = -3.4e38f; }
    __syncthreads();
  }
}

// scores + softmax + weighted-V update, scattered into CTX at selected rows
__global__ __launch_bounds__(64) void prob_scores_kernel(
    const float* __restrict__ Qb, const float* __restrict__ Kb,
    const float* __restrict__ Vb, const int* __restrict__ Mtop,
    float* __restrict__ CTX, float* __restrict__ SC,
    int Lq, int Lk, int u, int mask_flag)
{
  const int bh = blockIdx.x / u, ui = blockIdx.x % u;
  const int b = bh / N_HEADS, h = bh % N_HEADS;
  const int t = threadIdx.x;
  const int qrow = Mtop[bh * u + ui];
  const int kmax = mask_flag ? qrow : (Lk - 1);
  float* sc = SC + (size_t)blockIdx.x * Lk;

  __shared__ float qv[D_HEAD];
  __shared__ float red[64];
  qv[t] = Qb[((size_t)b * Lq + qrow) * D_MODEL + h * D_HEAD + t];
  __syncthreads();

  float lmax = -3.4e38f;
  for (int k = t; k <= kmax; k += 64) {
    float s = 0.f;
#pragma unroll
    for (int d = 0; d < D_HEAD; ++d)
      s += qv[d] * Kb[((size_t)b * Lk + k) * D_MODEL + h * D_HEAD + d];
    s *= 0.125f;                       // 1/sqrt(64)
    sc[k] = s;
    lmax = fmaxf(lmax, s);
  }
  red[t] = lmax;
  for (int off = 32; off > 0; off >>= 1) {
    __syncthreads();
    if (t < off) red[t] = fmaxf(red[t], red[t + off]);
  }
  __syncthreads();
  const float mx = red[0];
  __syncthreads();

  float lsum = 0.f;
  for (int k = t; k <= kmax; k += 64) lsum += __expf(sc[k] - mx);
  red[t] = lsum;
  for (int off = 32; off > 0; off >>= 1) {
    __syncthreads();
    if (t < off) red[t] += red[t + off];
  }
  __syncthreads();
  const float denom = red[0];

  float acc = 0.f;
  for (int k = 0; k <= kmax; ++k)
    acc += __expf(sc[k] - mx) * Vb[((size_t)b * Lk + k) * D_MODEL + h * D_HEAD + t];
  CTX[((size_t)b * Lq + qrow) * D_MODEL + h * D_HEAD + t] = acc / denom;
}

// ---------------------------------------------------------------------------
// Final projection: out[b, t, c] from last PRED_LEN decoder rows
// ---------------------------------------------------------------------------
__global__ void proj_kernel(const float* __restrict__ dec,
                            const float* __restrict__ pw,
                            const float* __restrict__ pb,
                            float* __restrict__ out)
{
  const int idx = blockIdx.x * blockDim.x + threadIdx.x;
  const int total = BATCH * PRED_LEN * C_OUT;
  if (idx >= total) return;
  const int c = idx % C_OUT;
  const int t = (idx / C_OUT) % PRED_LEN;
  const int b = idx / (C_OUT * PRED_LEN);
  const int l = (L_DEC - PRED_LEN) + t;
  float s = pb[c];
  for (int d = 0; d < D_MODEL; ++d)
    s += dec[((size_t)b * L_DEC + l) * D_MODEL + d] * pw[c * D_MODEL + d];
  out[idx] = s;
}

// ---------------------------------------------------------------------------
// Host orchestration
// ---------------------------------------------------------------------------
static inline void run_gemm(const float* A, const float* W, const float* bias,
                            float* C, int M, int N, int K, int act,
                            hipStream_t s) {
  dim3 grid(N / BN, M / BM);
  if (act)
    wmma_gemm_kernel<1><<<grid, 256, 0, s>>>(A, W, bias, C, M, N, K);
  else
    wmma_gemm_kernel<0><<<grid, 256, 0, s>>>(A, W, bias, C, M, N, K);
}

static inline void run_ln(const float* X, const float* R, const float* g,
                          const float* b, float* out, int rows, hipStream_t s) {
  layernorm_kernel<<<rows / 8, 256, 0, s>>>(X, R, g, b, out, rows);
}

struct AttnW { const float *wq,*bq,*wk,*bk,*wv,*bv,*wo,*bo; };

static void run_attn(const float* xq, int Mq, int Lq,
                     const float* xkv, int Mk, int Lk,
                     const AttnW& w, int mask_flag, unsigned salt,
                     float* Qb, float* Kb, float* Vb, float* CTX, float* T1,
                     float* SC, int* MTOP, hipStream_t s)
{
  run_gemm(xq,  w.wq, w.bq, Qb, Mq, D_MODEL, D_MODEL, 0, s);
  run_gemm(xkv, w.wk, w.bk, Kb, Mk, D_MODEL, D_MODEL, 0, s);
  run_gemm(xkv, w.wv, w.bv, Vb, Mk, D_MODEL, D_MODEL, 0, s);

  const int Upart = (FACTOR * (int)std::ceil(std::log((double)Lk)) < Lk)
                        ? FACTOR * (int)std::ceil(std::log((double)Lk)) : Lk;
  const int u = (FACTOR * (int)std::ceil(std::log((double)Lq)) < Lq)
                    ? FACTOR * (int)std::ceil(std::log((double)Lq)) : Lq;

  if (mask_flag)
    ctx_cumsum_kernel<<<BATCH * N_HEADS, 64, 0, s>>>(Vb, CTX, Lq);
  else
    ctx_mean_kernel<<<BATCH * N_HEADS, 256, 0, s>>>(Vb, CTX, Lq, Lk);

  prob_topk_kernel<<<BATCH * N_HEADS, 256, 0, s>>>(Qb, Kb, MTOP, Lq, Lk,
                                                   Upart, u, salt);
  prob_scores_kernel<<<BATCH * N_HEADS * u, 64, 0, s>>>(Qb, Kb, Vb, MTOP, CTX,
                                                        SC, Lq, Lk, u, mask_flag);

  run_gemm(CTX, w.wo, w.bo, T1, Mq, D_MODEL, D_MODEL, 0, s);
}

extern "C" void kernel_launch(void* const* d_in, const int* in_sizes, int n_in,
                              void* d_out, int out_size, void* d_ws, size_t ws_size,
                              hipStream_t stream) {
  auto F = [&](int i) { return (const float*)d_in[i]; };
  auto I = [&](int i) { return (const int*)d_in[i]; };

  // Input layout (setup_inputs insertion order, params flattened in dict order):
  // 0 x_state_enc, 1 x_enc, 2 x_mark_enc(unused), 3 x_state_dec, 4 x_dec,
  // 5 x_mark_dec(unused),
  // 6 enc_conv_w, 7 enc_state_emb, 8 dec_conv_w, 9 dec_state_emb,
  // enc layer i (i=0,1) at 10+16*i: wq,bq,wk,bk,wv,bv,wo,bo,w1,b1,w2,b2,
  //                                 ln1_g,ln1_b,ln2_g,ln2_b
  // dec layer at 42: self_attn(8)=42..49, cross_attn(8)=50..57,
  //                  w1=58,b1=59,w2=60,b2=61, ln1_g..ln3_b = 62..67
  // 68 enc_norm_g, 69 enc_norm_b, 70 dec_norm_g, 71 dec_norm_b,
  // 72 proj_w, 73 proj_b
  const int ENC_L0 = 10, ENC_L1 = 26, DEC_L = 42;

  // workspace layout (floats)
  float* ws = (float*)d_ws;
  const size_t SZ_ENC = (size_t)BATCH * L_ENC * D_MODEL;   // 4,194,304
  const size_t SZ_DEC = (size_t)BATCH * L_DEC * D_MODEL;   // 2,097,152
  float* enc_x = ws;                         // SZ_ENC
  float* dec_x = enc_x + SZ_ENC;             // SZ_DEC
  float* Qb    = dec_x + SZ_DEC;             // SZ_ENC (max rows)
  float* Kb    = Qb + SZ_ENC;
  float* Vb    = Kb + SZ_ENC;
  float* CTX   = Vb + SZ_ENC;
  float* T1    = CTX + SZ_ENC;
  float* T0    = T1 + SZ_ENC;                // FFN hidden: 8192*2048
  float* SC    = T0 + (size_t)BATCH * L_ENC * D_FF;
  int*   MTOP  = (int*)(SC + (size_t)BATCH * N_HEADS * 40 * 2048);

  const int ENC_M = BATCH * L_ENC;   // 8192
  const int DEC_M = BATCH * L_DEC;   // 4096

  // ---------------- encoder ----------------
  {
    const int total = ENC_M * D_MODEL;
    embed_kernel<<<(total + 255) / 256, 256, 0, stream>>>(
        I(0), F(1), F(6), F(7), enc_x, L_ENC);
  }
  for (int li = 0; li < 2; ++li) {
    const int p = (li == 0) ? ENC_L0 : ENC_L1;
    AttnW aw{F(p+0), F(p+1), F(p+2), F(p+3), F(p+4), F(p+5), F(p+6), F(p+7)};
    run_attn(enc_x, ENC_M, L_ENC, enc_x, ENC_M, L_ENC, aw, 0, (unsigned)li,
             Qb, Kb, Vb, CTX, T1, SC, MTOP, stream);
    run_ln(enc_x, T1, F(p+12), F(p+13), enc_x, ENC_M, stream);      // ln1(x+attn)
    run_gemm(enc_x, F(p+8),  F(p+9),  T0, ENC_M, D_FF,    D_MODEL, 1, stream);
    run_gemm(T0,    F(p+10), F(p+11), T1, ENC_M, D_MODEL, D_FF,    0, stream);
    run_ln(enc_x, T1, F(p+14), F(p+15), enc_x, ENC_M, stream);      // ln2(x+ffn)
  }
  run_ln(enc_x, nullptr, F(68), F(69), enc_x, ENC_M, stream);       // enc_norm

  // ---------------- decoder ----------------
  {
    const int total = DEC_M * D_MODEL;
    embed_kernel<<<(total + 255) / 256, 256, 0, stream>>>(
        I(3), F(4), F(8), F(9), dec_x, L_DEC);
  }
  {
    const int p = DEC_L;
    AttnW sa{F(p+0), F(p+1), F(p+2), F(p+3), F(p+4), F(p+5), F(p+6), F(p+7)};
    AttnW ca{F(p+8), F(p+9), F(p+10), F(p+11), F(p+12), F(p+13), F(p+14), F(p+15)};
    // self attention (masked, causal)
    run_attn(dec_x, DEC_M, L_DEC, dec_x, DEC_M, L_DEC, sa, 1, 2u,
             Qb, Kb, Vb, CTX, T1, SC, MTOP, stream);
    run_ln(dec_x, T1, F(p+20), F(p+21), dec_x, DEC_M, stream);      // ln1
    // cross attention (keys/values from encoder output)
    run_attn(dec_x, DEC_M, L_DEC, enc_x, ENC_M, L_ENC, ca, 0, 3u,
             Qb, Kb, Vb, CTX, T1, SC, MTOP, stream);
    run_ln(dec_x, T1, F(p+22), F(p+23), dec_x, DEC_M, stream);      // ln2
    // FFN
    run_gemm(dec_x, F(p+16), F(p+17), T0, DEC_M, D_FF,    D_MODEL, 1, stream);
    run_gemm(T0,    F(p+18), F(p+19), T1, DEC_M, D_MODEL, D_FF,    0, stream);
    run_ln(dec_x, T1, F(p+24), F(p+25), dec_x, DEC_M, stream);      // ln3
  }
  run_ln(dec_x, nullptr, F(70), F(71), dec_x, DEC_M, stream);       // dec_norm

  // ---------------- output projection ----------------
  {
    const int total = BATCH * PRED_LEN * C_OUT;
    proj_kernel<<<(total + 255) / 256, 256, 0, stream>>>(
        dec_x, F(72), F(73), (float*)d_out);
  }
}
